// xLSTMEncoder_6270652252571
// MI455X (gfx1250) — compile-verified, matrix operand order checked by reading, not run on hardware
//
#include <hip/hip_runtime.h>

// ---------------- types / constants ----------------
typedef __attribute__((ext_vector_type(16))) __bf16 v16bf;
typedef __attribute__((ext_vector_type(8)))  __bf16 v8bf;   // 16 bytes
typedef __attribute__((ext_vector_type(8)))  float  v8f;

constexpr int CB    = 2;      // batch
constexpr int CS    = 2048;   // seq len
constexpr int CIN   = 512;
constexpr int CE    = 1024;
constexpr int CINNER= 2048;
constexpr int CNH   = 4;
constexpr int CDH   = 512;
constexpr int CM    = CB * CS;     // 4096 token rows
constexpr float RSQRT_DH = 0.044194173824159216f; // 1/sqrt(512)
constexpr float NEG_INF  = -1e30f;

// ---------------- WMMA fragment loader: bf16 in memory, K-contiguous ----------------
// lane&15 selects row(A)/col(B); p = base + kb where kb = chunk + (lane>>4)*8.
// elems 0..7 -> K=kb+0..7 (contig 16B), elems 8..15 -> K=kb+16..23 (contig 16B).
__device__ __forceinline__ v16bf load_frag(const __bf16* __restrict__ p) {
  v8bf lo = *(const v8bf*)(p);
  v8bf hi = *(const v8bf*)(p + 16);
  v16bf r;
#pragma unroll
  for (int e = 0; e < 8; ++e) { r[e] = lo[e]; r[e + 8] = hi[e]; }
  return r;
}

// ---------------- conversion / transpose kernels (run once per launch) ----------------
__global__ void cvt_bf16_kernel(const float* __restrict__ X, __bf16* __restrict__ Y, size_t n) {
  size_t i = (size_t)blockIdx.x * 256 + threadIdx.x;
  if (i < n) Y[i] = (__bf16)X[i];
}
// X is [K][N] row-major f32 -> Y is [N][K] row-major bf16 (N-major for B-fragments)
__global__ void transpose_cvt_kernel(const float* __restrict__ X, __bf16* __restrict__ Y,
                                     int K, int N) {
  size_t i = (size_t)blockIdx.x * 256 + threadIdx.x;
  if (i >= (size_t)K * N) return;
  int k = (int)(i / N), n = (int)(i - (size_t)k * N);
  Y[(size_t)n * K + k] = (__bf16)X[i];
}

// ---------------- generic WMMA GEMM: C = A[MxK]bf16 @ Bt[NxK]bf16 + bias (+ res) ----------------
// one wave -> 32x32 output tile (2x2 wmma tiles); M,N,K multiples of 32.
__global__ void gemm_bias_kernel(const __bf16* __restrict__ A, const __bf16* __restrict__ Bt,
                                 const float* __restrict__ bias, const float* __restrict__ res,
                                 float* __restrict__ C, int M, int N, int K) {
  int gw   = (int)(((size_t)blockIdx.x * blockDim.x + threadIdx.x) >> 5);
  int lane = threadIdx.x & 31;
  int half = lane >> 4, l16 = lane & 15;
  int tilesN = N >> 5;
  int tr = gw / tilesN;
  int tm = tr << 5;
  int tn = (gw - tr * tilesN) << 5;
  if (tm >= M) return;

  const __bf16* ar0 = A  + (size_t)(tm + l16)      * K;
  const __bf16* ar1 = A  + (size_t)(tm + 16 + l16) * K;
  const __bf16* bc0 = Bt + (size_t)(tn + l16)      * K;
  const __bf16* bc1 = Bt + (size_t)(tn + 16 + l16) * K;

  v8f acc00 = {0,0,0,0,0,0,0,0}, acc01 = {0,0,0,0,0,0,0,0};
  v8f acc10 = {0,0,0,0,0,0,0,0}, acc11 = {0,0,0,0,0,0,0,0};

  for (int k0 = 0; k0 < K; k0 += 32) {
    int kb = k0 + half * 8;
    if (k0 + 32 < K) {                       // prefetch next K-chunk (global_prefetch_b8)
      __builtin_prefetch(ar0 + kb + 32, 0, 1);
      __builtin_prefetch(bc0 + kb + 32, 0, 1);
    }
    v16bf a0 = load_frag(ar0 + kb);
    v16bf a1 = load_frag(ar1 + kb);
    v16bf b0 = load_frag(bc0 + kb);
    v16bf b1 = load_frag(bc1 + kb);
    acc00 = __builtin_amdgcn_wmma_f32_16x16x32_bf16(false, a0, false, b0, (short)0, acc00, false, false);
    acc01 = __builtin_amdgcn_wmma_f32_16x16x32_bf16(false, a0, false, b1, (short)0, acc01, false, false);
    acc10 = __builtin_amdgcn_wmma_f32_16x16x32_bf16(false, a1, false, b0, (short)0, acc10, false, false);
    acc11 = __builtin_amdgcn_wmma_f32_16x16x32_bf16(false, a1, false, b1, (short)0, acc11, false, false);
  }

  int c0 = tn + l16, c1 = tn + 16 + l16;
  float bi0 = bias[c0], bi1 = bias[c1];
#pragma unroll
  for (int r = 0; r < 8; ++r) {
    int r0 = tm + half * 8 + r;
    int r1 = r0 + 16;
    size_t i00 = (size_t)r0 * N + c0, i01 = (size_t)r0 * N + c1;
    size_t i10 = (size_t)r1 * N + c0, i11 = (size_t)r1 * N + c1;
    float v00 = acc00[r] + bi0, v01 = acc01[r] + bi1;
    float v10 = acc10[r] + bi0, v11 = acc11[r] + bi1;
    if (res) { v00 += res[i00]; v01 += res[i01]; v10 += res[i10]; v11 += res[i11]; }
    C[i00] = v00; C[i01] = v01; C[i10] = v10; C[i11] = v11;
  }
}

// ---------------- LayerNorm (block per row); optional f32 and/or bf16 outputs ----------------
__global__ void layernorm_kernel(const float* __restrict__ X, const float* __restrict__ w,
                                 float* __restrict__ Yf, __bf16* __restrict__ Yb, int N) {
  int row = blockIdx.x;
  int tid = threadIdx.x;
  const float* xr = X + (size_t)row * N;
  float s = 0.f, ss = 0.f;
  for (int i = tid; i < N; i += 256) { float v = xr[i]; s += v; ss += v * v; }
  __shared__ float rs[256], rq[256];
  rs[tid] = s; rq[tid] = ss; __syncthreads();
  for (int off = 128; off > 0; off >>= 1) {
    if (tid < off) { rs[tid] += rs[tid + off]; rq[tid] += rq[tid + off]; }
    __syncthreads();
  }
  float mean = rs[0] / N;
  float var  = rq[0] / N - mean * mean;
  float rstd = rsqrtf(var + 1e-5f);
  for (int i = tid; i < N; i += 256) {
    float y = (xr[i] - mean) * rstd * w[i];
    if (Yf) Yf[(size_t)row * N + i] = y;
    if (Yb) Yb[(size_t)row * N + i] = (__bf16)y;
  }
}

// ---------------- causal depthwise conv1d (K=4) + SiLU ----------------
__global__ void conv_silu_kernel(const float* __restrict__ up, const float* __restrict__ cw,
                                 const float* __restrict__ cb, float* __restrict__ xca) {
  size_t idx = (size_t)blockIdx.x * 256 + threadIdx.x;  // over CM*CINNER
  int c = (int)(idx & (CINNER - 1));
  size_t row = idx >> 11;
  int s = (int)(row & (CS - 1));
  float acc = cb[c];
#pragma unroll
  for (int t = 0; t < 4; ++t) {
    int sp = s - 3 + t;
    if (sp >= 0) acc += cw[c * 4 + t] * up[(row + (sp - s)) * (size_t)(2 * CINNER) + c];
  }
  float sg = 1.f / (1.f + __expf(-acc));
  xca[idx] = acc * sg;
}

// ---------------- headwise (block-diagonal 4x4) q/k/v -> bf16 (v transposed per head) ----------------
__global__ void qkv_kernel(const float* __restrict__ xca, const float* __restrict__ up,
                           const float* __restrict__ qw, const float* __restrict__ qb,
                           const float* __restrict__ kw, const float* __restrict__ kb,
                           const float* __restrict__ vw, const float* __restrict__ vb,
                           __bf16* __restrict__ qo, __bf16* __restrict__ ko,
                           __bf16* __restrict__ vt) {
  size_t idx = (size_t)blockIdx.x * 256 + threadIdx.x;  // over CM*CINNER
  int c = (int)(idx & (CINNER - 1));
  size_t row = idx >> 11;
  int h = c >> 2, o = c & 3;
  float aq = qb[c], ak = kb[c], av = vb[c];
  const float* xc = xca + row * CINNER + h * 4;
  const float* xm = up  + row * (size_t)(2 * CINNER) + h * 4;
#pragma unroll
  for (int d = 0; d < 4; ++d) {
    int wi = h * 16 + o * 4 + d;
    aq += qw[wi] * xc[d];
    ak += kw[wi] * xc[d];
    av += vw[wi] * xm[d];
  }
  qo[idx] = (__bf16)aq;
  ko[idx] = (__bf16)ak;
  // v transposed: [bnh][dh][s]
  int b = (int)(row >> 11);           // row / CS
  int s = (int)(row & (CS - 1));
  int nh = c >> 9, dh = c & (CDH - 1);
  vt[(((size_t)(b * CNH + nh)) * CDH + dh) * CS + s] = (__bf16)av;
}

// ---------------- input/forget gate matvecs + log-sigmoid ----------------
__global__ void gates_kernel(const __bf16* __restrict__ q, const __bf16* __restrict__ k,
                             const __bf16* __restrict__ vt,
                             const float* __restrict__ igw, const float* __restrict__ igb,
                             const float* __restrict__ fgw, const float* __restrict__ fgb,
                             float* __restrict__ ig_out, float* __restrict__ logf_out) {
  int row = blockIdx.x;                   // token
  int tid = threadIdx.x;
  int b = row / CS, s = row - b * CS;
  float aI[4] = {0,0,0,0}, aF[4] = {0,0,0,0};
  size_t base = (size_t)row * CINNER;
  for (int f = tid; f < 3 * CINNER; f += 256) {
    float val;
    if (f < CINNER)          val = (float)q[base + f];
    else if (f < 2 * CINNER) val = (float)k[base + f - CINNER];
    else {
      int c = f - 2 * CINNER;
      int nh = c >> 9, dh = c & (CDH - 1);
      val = (float)vt[(((size_t)(b * CNH + nh)) * CDH + dh) * CS + s];
    }
#pragma unroll
    for (int h = 0; h < 4; ++h) {
      aI[h] += val * igw[(size_t)f * 4 + h];
      aF[h] += val * fgw[(size_t)f * 4 + h];
    }
  }
  __shared__ float red[256];
  for (int ch = 0; ch < 8; ++ch) {
    red[tid] = (ch < 4) ? aI[ch] : aF[ch - 4];
    __syncthreads();
    for (int off = 128; off > 0; off >>= 1) {
      if (tid < off) red[tid] += red[tid + off];
      __syncthreads();
    }
    if (tid == 0) {
      int h = ch & 3;
      size_t oi = ((size_t)(b * CNH + h)) * CS + s;
      if (ch < 4) ig_out[oi] = red[0] + igb[h];
      else {
        float x = red[0] + fgb[h];
        logf_out[oi] = fminf(x, 0.f) - log1pf(__expf(-fabsf(x)));  // log_sigmoid
      }
    }
    __syncthreads();
  }
}

// ---------------- inclusive cumsum of log_f over S, per (b,nh) ----------------
__global__ void cumsum_kernel(const float* __restrict__ logf, float* __restrict__ cum) {
  int t = threadIdx.x;
  if (t < CB * CNH) {
    size_t base = (size_t)t * CS;
    float r = 0.f;
    for (int s = 0; s < CS; ++s) { r += logf[base + s]; cum[base + s] = r; }
  }
}

// ---------------- mLSTM flash-style kernel ----------------
// grid: (S/16, B*NH); block 256 = 8 waves; wave w owns DH slice [w*64, w*64+64)
__global__ void mlstm_kernel(const __bf16* __restrict__ q, const __bf16* __restrict__ k,
                             const __bf16* __restrict__ vt,
                             const float* __restrict__ ig, const float* __restrict__ cum,
                             const float* __restrict__ up, const float* __restrict__ xca,
                             const float* __restrict__ onw, const float* __restrict__ skp,
                             __bf16* __restrict__ hg) {
  const int iblk = blockIdx.x;
  const int bnh  = blockIdx.y;
  const int b    = bnh >> 2, nh = bnh & 3;
  const int i0   = iblk * 16;
  const int tid  = threadIdx.x;
  const int wv   = tid >> 5;
  const int lane = tid & 31;
  const int half = lane >> 4, l16 = lane & 15;

  __shared__ float  sc_part[8][16][16];
  __shared__ float  s_logD[16][16];
  __shared__ float  sP[16][16];
  __shared__ __bf16 sPb[16][16];           // bf16 copy for PV A-fragment
  __shared__ float  h_sm[16][CDH];
  __shared__ float  red_s[16][16], red_q[16][16];
  __shared__ float  m_row[16], s_row[16], scl[16], cum_i[16], cum_j[16], ig_j[16];
  __shared__ float  denom_r[16], mean_r[16], rstd_r[16];

  const __bf16* qh = q + (size_t)nh * CDH;
  const __bf16* kh = k + (size_t)nh * CDH;

  if (tid < 16) {
    m_row[tid] = NEG_INF;
    s_row[tid] = 0.f;
    cum_i[tid] = cum[(size_t)bnh * CS + i0 + tid];
  }
  __syncthreads();

  // preload q A-fragments for this wave's DH slice (2 x 32-wide K chunks)
  v16bf qfrag[2];
#pragma unroll
  for (int t = 0; t < 2; ++t)
    qfrag[t] = load_frag(qh + ((size_t)(b * CS) + i0 + l16) * CINNER + wv * 64 + t * 32 + half * 8);

  v8f hacc[4];
#pragma unroll
  for (int n = 0; n < 4; ++n) hacc[n] = (v8f){0,0,0,0,0,0,0,0};

  for (int jb = 0; jb <= iblk; ++jb) {
    const int j0 = jb * 16;
    if (tid < 16) {
      cum_j[tid] = cum[(size_t)bnh * CS + j0 + tid];
      ig_j[tid]  = ig [(size_t)bnh * CS + j0 + tid];
    }
    if (jb < iblk) {   // prefetch next j-block's k row and v column (global_prefetch_b8)
      __builtin_prefetch(kh + ((size_t)(b * CS) + j0 + 16 + l16) * CINNER + wv * 64 + half * 8, 0, 1);
      __builtin_prefetch(vt + (((size_t)bnh) * CDH + wv * 64 + l16) * CS + j0 + 16, 0, 1);
    }
    // --- partial scores: this wave's 64-wide DH slice ---
    v8f acc = {0,0,0,0,0,0,0,0};
#pragma unroll
    for (int t = 0; t < 2; ++t) {
      v16bf kf = load_frag(kh + ((size_t)(b * CS) + j0 + l16) * CINNER + wv * 64 + t * 32 + half * 8);
      acc = __builtin_amdgcn_wmma_f32_16x16x32_bf16(false, qfrag[t], false, kf, (short)0, acc, false, false);
    }
#pragma unroll
    for (int r = 0; r < 8; ++r) sc_part[wv][r + half * 8][l16] = acc[r];
    __syncthreads();

    // --- combine partials + decay matrix ---
    {
      int row = tid >> 4, col = tid & 15;
      float qk = 0.f;
#pragma unroll
      for (int w = 0; w < 8; ++w) qk += sc_part[w][row][col];
      qk *= RSQRT_DH;
      sc_part[0][row][col] = qk;
      bool valid = (jb < iblk) || (col <= row);
      s_logD[row][col] = valid ? (cum_i[row] - cum_j[col] + ig_j[col]) : NEG_INF;
    }
    __syncthreads();

    // --- online max / rescale factors ---
    if (tid < 16) {
      float mb = NEG_INF;
#pragma unroll
      for (int c = 0; c < 16; ++c) mb = fmaxf(mb, s_logD[tid][c]);
      float mold = m_row[tid];
      float mnew = fmaxf(mold, mb);
      scl[tid]   = __expf(mold - mnew);
      m_row[tid] = mnew;
      s_row[tid] *= scl[tid];
    }
    __syncthreads();

    // --- P = qk * exp(logD - m) (f32 for row-sum, bf16 for PV A-frag) ---
    {
      int row = tid >> 4, col = tid & 15;
      float p = sc_part[0][row][col] * __expf(s_logD[row][col] - m_row[row]);
      sP[row][col]  = p;
      sPb[row][col] = (__bf16)p;
    }
    __syncthreads();
    if (tid < 16) {
      float ssum = 0.f;
#pragma unroll
      for (int c = 0; c < 16; ++c) ssum += sP[tid][c];
      s_row[tid] += ssum;
    }
    __syncthreads();

    // --- rescale hacc, then hacc += P @ V_jblock (K padded 16->32) ---
    {
      float sc8[8];
#pragma unroll
      for (int r = 0; r < 8; ++r) sc8[r] = scl[r + half * 8];
#pragma unroll
      for (int n = 0; n < 4; ++n)
#pragma unroll
        for (int r = 0; r < 8; ++r) hacc[n][r] *= sc8[r];

      int pkb = half * 8;
      v8bf plo = *(const v8bf*)&sPb[l16][pkb];
      v16bf pa;
#pragma unroll
      for (int e = 0; e < 8; ++e) { pa[e] = plo[e]; pa[e + 8] = (__bf16)0.f; }

#pragma unroll
      for (int n = 0; n < 4; ++n) {
        int dh = wv * 64 + n * 16 + l16;
        v8bf vlo = *(const v8bf*)(vt + (((size_t)bnh) * CDH + dh) * CS + j0 + pkb);
        v16bf vb;
#pragma unroll
        for (int e = 0; e < 8; ++e) { vb[e] = vlo[e]; vb[e + 8] = (__bf16)0.f; }
        hacc[n] = __builtin_amdgcn_wmma_f32_16x16x32_bf16(false, pa, false, vb, (short)0, hacc[n], false, false);
      }
    }
    __syncthreads();
  }

  // --- epilogue: denom, group-norm over DH, onorm_w, skip, SiLU(z) gate ---
  if (tid < 16) denom_r[tid] = fmaxf(fabsf(s_row[tid]), __expf(-m_row[tid])) + 1e-6f;
  __syncthreads();
  {
    float dn[8];
#pragma unroll
    for (int r = 0; r < 8; ++r) dn[r] = denom_r[r + half * 8];
#pragma unroll
    for (int n = 0; n < 4; ++n)
#pragma unroll
      for (int r = 0; r < 8; ++r)
        h_sm[r + half * 8][wv * 64 + n * 16 + l16] = hacc[n][r] / dn[r];
  }
  __syncthreads();
  {
    int row = tid >> 4, sub = tid & 15;
    float s = 0.f, ss = 0.f;
#pragma unroll
    for (int e = 0; e < 32; ++e) { float x = h_sm[row][sub * 32 + e]; s += x; ss += x * x; }
    red_s[row][sub] = s; red_q[row][sub] = ss;
  }
  __syncthreads();
  if (tid < 16) {
    float s = 0.f, ss = 0.f;
#pragma unroll
    for (int c = 0; c < 16; ++c) { s += red_s[tid][c]; ss += red_q[tid][c]; }
    float mean = s / (float)CDH;
    float var  = ss / (float)CDH - mean * mean;
    mean_r[tid] = mean; rstd_r[tid] = rsqrtf(var + 1e-5f);
  }
  __syncthreads();
  {
    int row = tid >> 4, sub = tid & 15;
    size_t srow = (size_t)(b * CS) + i0 + row;
    for (int e = 0; e < 32; ++e) {
      int dh = sub * 32 + e;
      int f  = nh * CDH + dh;
      float hn = (h_sm[row][dh] - mean_r[row]) * rstd_r[row] * onw[f];
      float xc = xca[srow * CINNER + f];
      float zz = up[srow * (size_t)(2 * CINNER) + CINNER + f];
      float sz = zz / (1.f + __expf(-zz));
      hg[srow * CINNER + f] = (__bf16)((hn + skp[f] * xc) * sz);
    }
  }
}

// ---------------- host launch ----------------
extern "C" void kernel_launch(void* const* d_in, const int* in_sizes, int n_in,
                              void* d_out, int out_size, void* d_ws, size_t ws_size,
                              hipStream_t stream) {
  const float* x      = (const float*)d_in[0];
  const float* w_in   = (const float*)d_in[1];
  const float* b_in   = (const float*)d_in[2];
  const float* ln1_w  = (const float*)d_in[3];
  const float* w_up   = (const float*)d_in[4];
  const float* b_up   = (const float*)d_in[5];
  const float* conv_w = (const float*)d_in[6];
  const float* conv_b = (const float*)d_in[7];
  const float* q_w    = (const float*)d_in[8];
  const float* q_b    = (const float*)d_in[9];
  const float* k_w    = (const float*)d_in[10];
  const float* k_b    = (const float*)d_in[11];
  const float* v_w    = (const float*)d_in[12];
  const float* v_b    = (const float*)d_in[13];
  const float* ig_w   = (const float*)d_in[14];
  const float* ig_b   = (const float*)d_in[15];
  const float* fg_w   = (const float*)d_in[16];
  const float* fg_b   = (const float*)d_in[17];
  const float* onorm_w= (const float*)d_in[18];
  const float* skip   = (const float*)d_in[19];
  const float* w_down = (const float*)d_in[20];
  const float* b_down = (const float*)d_in[21];
  const float* post_w = (const float*)d_in[22];

  // 256B-aligned workspace allocator
  char*  base = (char*)d_ws;
  size_t off  = 0;
  auto alloc = [&](size_t bytes) -> void* {
    void* p = base + off;
    off += (bytes + 255) & ~(size_t)255;
    return p;
  };
  float*  h_in = (float*) alloc((size_t)CM * CE * 4);
  float*  up   = (float*) alloc((size_t)CM * 2 * CINNER * 4);
  float*  xca  = (float*) alloc((size_t)CM * CINNER * 4);
  float*  tbuf = (float*) alloc((size_t)CM * CE * 4);
  float*  igb  = (float*) alloc((size_t)CB * CNH * CS * 4);
  float*  logf = (float*) alloc((size_t)CB * CNH * CS * 4);
  float*  cumb = (float*) alloc((size_t)CB * CNH * CS * 4);
  __bf16* xbf  = (__bf16*)alloc((size_t)CM * CIN * 2);
  __bf16* xnbf = (__bf16*)alloc((size_t)CM * CE * 2);
  __bf16* hgbf = (__bf16*)alloc((size_t)CM * CINNER * 2);
  __bf16* qbf  = (__bf16*)alloc((size_t)CM * CINNER * 2);
  __bf16* kbf  = (__bf16*)alloc((size_t)CM * CINNER * 2);
  __bf16* vtbf = (__bf16*)alloc((size_t)CM * CINNER * 2);
  __bf16* winT = (__bf16*)alloc((size_t)CE * CIN * 2);
  __bf16* wupT = (__bf16*)alloc((size_t)(2 * CINNER) * CE * 2);
  __bf16* wdnT = (__bf16*)alloc((size_t)CE * CINNER * 2);

  auto blocks_for = [](size_t n) { return (int)((n + 255) / 256); };
  auto gemm_blocks = [](int M, int N) { return (M / 32) * (N / 32) * 32 / 256; };

  // 0) one-time converts: activations x -> bf16; weights -> bf16 + N-major transpose
  cvt_bf16_kernel<<<blocks_for((size_t)CM * CIN), 256, 0, stream>>>(x, xbf, (size_t)CM * CIN);
  transpose_cvt_kernel<<<blocks_for((size_t)CIN * CE), 256, 0, stream>>>(w_in,  winT, CIN, CE);
  transpose_cvt_kernel<<<blocks_for((size_t)CE * 2 * CINNER), 256, 0, stream>>>(w_up, wupT, CE, 2 * CINNER);
  transpose_cvt_kernel<<<blocks_for((size_t)CINNER * CE), 256, 0, stream>>>(w_down, wdnT, CINNER, CE);

  // 1) h_in = x @ w_in + b_in
  gemm_bias_kernel<<<gemm_blocks(CM, CE), 256, 0, stream>>>(xbf, winT, b_in, nullptr, h_in, CM, CE, CIN);
  // 2) xn = LN(h_in) * ln1_w  (bf16 only; feeds gemm2)
  layernorm_kernel<<<CM, 256, 0, stream>>>(h_in, ln1_w, nullptr, xnbf, CE);
  // 3) up = xn @ w_up + b_up
  gemm_bias_kernel<<<gemm_blocks(CM, 2 * CINNER), 256, 0, stream>>>(xnbf, wupT, b_up, nullptr, up,
                                                                    CM, 2 * CINNER, CE);
  // 4) causal conv + SiLU
  conv_silu_kernel<<<blocks_for((size_t)CM * CINNER), 256, 0, stream>>>(up, conv_w, conv_b, xca);
  // 5) headwise q/k/v (q,k bf16 row-major; v bf16 transposed per head)
  qkv_kernel<<<blocks_for((size_t)CM * CINNER), 256, 0, stream>>>(xca, up, q_w, q_b, k_w, k_b, v_w, v_b,
                                                                  qbf, kbf, vtbf);
  // 6) gates + log-sigmoid
  gates_kernel<<<CM, 256, 0, stream>>>(qbf, kbf, vtbf, ig_w, ig_b, fg_w, fg_b, igb, logf);
  // 7) cumsum of log_f
  cumsum_kernel<<<1, 32, 0, stream>>>(logf, cumb);
  // 8) mLSTM + head-norm + skip + gate -> bf16
  mlstm_kernel<<<dim3(CS / 16, CB * CNH), 256, 0, stream>>>(qbf, kbf, vtbf, igb, cumb, up, xca,
                                                            onorm_w, skip, hgbf);
  // 9) tbuf = hg @ w_down + b_down + h_in  (fused residual)
  gemm_bias_kernel<<<gemm_blocks(CM, CE), 256, 0, stream>>>(hgbf, wdnT, b_down, h_in, tbuf, CM, CE, CINNER);
  // 10) out = LN(tbuf) * post_w
  layernorm_kernel<<<CM, 256, 0, stream>>>(tbuf, post_w, (float*)d_out, nullptr, CE);
}
